// GaborNet_21431886807802
// MI455X (gfx1250) — matrix operand order repaired and checked by reading, not compiled, not run on hardware
//
#include <hip/hip_runtime.h>

typedef _Float16 h8   __attribute__((ext_vector_type(8)));
typedef _Float16 v16h __attribute__((ext_vector_type(16)));
typedef float    v8f  __attribute__((ext_vector_type(8)));

#define HWD    512
#define NPIX   (HWD * HWD)          // 262144 pixels per image
#define NBATCH 2
#define HID    64
#define NLAY   4
#define NFILT  (NLAY + 1)
#define OUTC   3
#define WAVES  4                    // waves per block (wave32)
#define TPB    (WAVES * 32)
#define TILEP  16                   // pixels per wave tile (WMMA M)
#define APAD   72                   // padded f16 row stride (bank-conflict break)

struct __align__(16) Smem {
  float4   gbA[NFILT * HID];            // A00, A01, t0, w0   per (filter,channel)
  float4   gbB[NFILT * HID];            // A10, A11, t1, w1
  float    gbP[NFILT * HID];            // sin phase offset b
  _Float16 w[NLAY][HID][APAD];          // lin_w as f16, [layer][o][i]
  float    linb[NLAY][HID];
  _Float16 outB[16][APAD];              // out_w padded to 16 cols, [n][k]
  float    outb[16];
  _Float16 act[WAVES][TILEP][APAD];     // per-wave activation tile [m][c]
  float    coords[WAVES][2][TILEP];     // per-wave pixel coords x0/x1
};

// Assemble a v16h from two contiguous 16B-aligned LDS runs (elements 0..7, 8..15).
__device__ __forceinline__ v16h load16h(const _Float16* p) {
  h8 lo = *(const h8*)(p);
  h8 hi = *(const h8*)(p + 8);
  v16h v;
#pragma unroll
  for (int e = 0; e < 8; ++e) { v[e] = lo[e]; v[e + 8] = hi[e]; }
  return v;
}

// A-matrix (16x32 f16) register image for this lane: two runs of 8 channels.
// lane<16: K{cs..cs+7, cs+16..cs+23}; p = act row of pixel m, cs = 32*chunk + 8*laneHi.
__device__ __forceinline__ v16h loadA(const _Float16* row, int cs) {
  h8 lo = *(const h8*)(row + cs);
  h8 hi = *(const h8*)(row + cs + 16);
  v16h v;
#pragma unroll
  for (int e = 0; e < 8; ++e) { v[e] = lo[e]; v[e + 8] = hi[e]; }
  return v;
}

// Gabor gate for one channel, 8 pixels (D-layout rows of this lane).
__device__ __forceinline__ void gabor8(const float4 pa, const float4 pb, const float ph,
                                       const float* px0, const float* px1, float* g) {
#pragma unroll
  for (int r = 0; r < 8; ++r) {
    float v0 = fmaf(pa.x, px0[r], fmaf(pa.y, px1[r], pa.z));
    float v1 = fmaf(pb.x, px0[r], fmaf(pb.y, px1[r], pb.z));
    float sa = fmaf(pa.w, px0[r], fmaf(pb.w, px1[r], ph));
    g[r] = __expf(-0.5f * fmaf(v0, v0, v1 * v1)) * __sinf(sa);
  }
}

__global__ __launch_bounds__(TPB) void gabor_fused(
    const float* __restrict__ x,  const float* __restrict__ fw,
    const float* __restrict__ fb, const float* __restrict__ mu,
    const float* __restrict__ ga, const float* __restrict__ th,
    const float* __restrict__ lw, const float* __restrict__ lb,
    const float* __restrict__ ow, const float* __restrict__ ob,
    float* __restrict__ out) {
  __shared__ Smem sm;
  const int tid = threadIdx.x;

  // ---------- per-block parameter staging ----------
  for (int i = tid; i < NFILT * HID; i += TPB) {
    float ang = 6.283185307179586f * th[i];
    float cc = __cosf(ang), ss = __sinf(ang);
    float g0 = ga[i * 2 + 0], g1 = ga[i * 2 + 1];
    float m0 = mu[i * 2 + 0], m1 = mu[i * 2 + 1];
    float A00 = g0 * cc, A01 = g0 * ss, A10 = -g1 * ss, A11 = g1 * cc;
    sm.gbA[i] = make_float4(A00, A01, -(A00 * m0 + A01 * m1), fw[i * 2 + 0]);
    sm.gbB[i] = make_float4(A10, A11, -(A10 * m0 + A11 * m1), fw[i * 2 + 1]);
    sm.gbP[i] = fb[i];
  }
  for (int i = tid; i < NLAY * HID * HID; i += TPB) {
    int l = i >> 12, rem = i & 4095, o = rem >> 6, k = rem & 63;
    sm.w[l][o][k] = (_Float16)lw[i];
  }
  for (int i = tid; i < NLAY * HID; i += TPB) ((float*)sm.linb)[i] = lb[i];
  for (int i = tid; i < 16 * HID; i += TPB) {
    int nn = i >> 6, k = i & 63;
    sm.outB[nn][k] = (nn < OUTC) ? (_Float16)ow[nn * HID + k] : (_Float16)0.0f;
  }
  if (tid < 16) sm.outb[tid] = (tid < OUTC) ? ob[tid] : 0.0f;
  __syncthreads();

  // ---------- per-wave 16-pixel tile ----------
  const int wave = tid >> 5;
  const int lane = tid & 31;
  const int hi   = lane >> 4;     // lane half
  const int nn   = lane & 15;     // WMMA N / M sub-index
  const int tile = blockIdx.x * WAVES + wave;
  const int p0   = tile * TILEP;
  const int b    = p0 / NPIX;
  const int r0   = p0 % NPIX;

  // Load the tile's 16 (x0,x1) coordinate pairs: half-wave per component.
  sm.coords[wave][hi][nn] = x[(size_t)(b * 2 + hi) * NPIX + r0 + nn];
  asm volatile("s_wait_dscnt 0x0" ::: "memory");

  // D-layout pixel rows for this lane: m = 8*hi + r.
  float px0[8], px1[8];
#pragma unroll
  for (int r = 0; r < 8; ++r) {
    px0[r] = sm.coords[wave][0][8 * hi + r];
    px1[r] = sm.coords[wave][1][8 * hi + r];
  }

  // Filter 0 -> activation tile (f16, [m][c]).
#pragma unroll
  for (int t = 0; t < 4; ++t) {
    int c = 16 * t + nn;
    float g[8];
    gabor8(sm.gbA[c], sm.gbB[c], sm.gbP[c], px0, px1, g);
#pragma unroll
    for (int r = 0; r < 8; ++r) sm.act[wave][8 * hi + r][c] = (_Float16)g[r];
  }
  asm volatile("s_wait_dscnt 0x0" ::: "memory");

#pragma unroll
  for (int l = 1; l <= NLAY; ++l) {
    // A = activations [16 x 64] as two 16x32 f16 register tiles.
    v16h a0 = loadA(&sm.act[wave][nn][0], 8 * hi);
    v16h a1 = loadA(&sm.act[wave][nn][0], 32 + 8 * hi);
    v8f acc[4];
#pragma unroll
    for (int t = 0; t < 4; ++t) {
      int c = 16 * t + nn;                 // output channel for this lane
      float bias = sm.linb[l - 1][c];
      v8f cacc;
#pragma unroll
      for (int r = 0; r < 8; ++r) cacc[r] = bias;
      v16h b0 = loadA(&sm.w[l - 1][c][0], 8 * hi);        // K 0..31
      cacc = __builtin_amdgcn_wmma_f32_16x16x32_f16(false, a0, false, b0,
                                                    (short)0, cacc, false, false);
      v16h b1 = loadA(&sm.w[l - 1][c][0], 32 + 8 * hi);   // K 32..63
      cacc = __builtin_amdgcn_wmma_f32_16x16x32_f16(false, a1, false, b1,
                                                    (short)0, cacc, false, false);
      acc[t] = cacc;
    }
    // Gate by gabor filter l, write back activation tile for next layer.
#pragma unroll
    for (int t = 0; t < 4; ++t) {
      int c = 16 * t + nn, pi = l * HID + c;
      float g[8];
      gabor8(sm.gbA[pi], sm.gbB[pi], sm.gbP[pi], px0, px1, g);
#pragma unroll
      for (int r = 0; r < 8; ++r)
        sm.act[wave][8 * hi + r][c] = (_Float16)(acc[t][r] * g[r]);
    }
    asm volatile("s_wait_dscnt 0x0" ::: "memory");
  }

  // Final 3x64 projection as one padded WMMA N-tile.
  v16h a0 = loadA(&sm.act[wave][nn][0], 8 * hi);
  v16h a1 = loadA(&sm.act[wave][nn][0], 32 + 8 * hi);
  float obias = sm.outb[nn];
  v8f facc;
#pragma unroll
  for (int r = 0; r < 8; ++r) facc[r] = obias;
  v16h fb0 = loadA(&sm.outB[nn][0], 8 * hi);
  facc = __builtin_amdgcn_wmma_f32_16x16x32_f16(false, a0, false, fb0,
                                                (short)0, facc, false, false);
  v16h fb1 = loadA(&sm.outB[nn][0], 32 + 8 * hi);
  facc = __builtin_amdgcn_wmma_f32_16x16x32_f16(false, a1, false, fb1,
                                                (short)0, facc, false, false);

  if (nn < OUTC) {
    size_t obase = (size_t)(b * OUTC + nn) * NPIX + r0 + 8 * hi;
#pragma unroll
    for (int r = 0; r < 8; ++r) out[obase + r] = facc[r];
  }
}

extern "C" void kernel_launch(void* const* d_in, const int* in_sizes, int n_in,
                              void* d_out, int out_size, void* d_ws, size_t ws_size,
                              hipStream_t stream) {
  const float* x  = (const float*)d_in[0];
  const float* fw = (const float*)d_in[1];
  const float* fb = (const float*)d_in[2];
  const float* mu = (const float*)d_in[3];
  const float* ga = (const float*)d_in[4];
  const float* th = (const float*)d_in[5];
  const float* lw = (const float*)d_in[6];
  const float* lb = (const float*)d_in[7];
  const float* ow = (const float*)d_in[8];
  const float* ob = (const float*)d_in[9];
  float* out = (float*)d_out;

  int total_pix = in_sizes[0] / 2;           // B*H*W (x is [B,2,H,W])
  int tiles  = total_pix / TILEP;            // 32768
  int blocks = tiles / WAVES;                // 8192
  if (blocks < 1) blocks = 1;
  gabor_fused<<<blocks, TPB, 0, stream>>>(x, fw, fb, mu, ga, th, lw, lb, ow, ob, out);
}
